// Qwen2VLVisionBlock_52432960750179
// MI455X (gfx1250) — matrix operand order repaired and checked by reading, not compiled, not run on hardware
//
#include <hip/hip_runtime.h>
#include <hip/hip_bf16.h>
#include <math.h>

#define DIM 1280
#define HEADS 16
#define HEAD_DIM 80
#define HDP 96            // padded head dim (3 x K=32 WMMA chunks)
#define MLP_HIDDEN 5120
#define S_LEN 2048
#define EPS 1e-6f

#define BM 128
#define BN 128
#define BK 32

typedef __attribute__((ext_vector_type(16))) _Float16 v16h;
typedef __attribute__((ext_vector_type(8)))  _Float16 v8h;
typedef __attribute__((ext_vector_type(8)))  float    v8f;
typedef __attribute__((ext_vector_type(4)))  int      v4i;

// addrspace-qualified pointee typedefs for the async-LDS builtin
typedef __attribute__((address_space(1))) v4i gv4i;  // global
typedef __attribute__((address_space(3))) v4i lv4i;  // LDS

#if __has_builtin(__builtin_amdgcn_global_load_async_to_lds_b128) && \
    __has_builtin(__builtin_amdgcn_s_wait_asynccnt)
#define USE_ASYNC_LDS 1
#else
#define USE_ASYNC_LDS 0
#endif

static __device__ __forceinline__ v8f wmma_f16(v16h a, v16h b, v8f c) {
  // D = A(16x32 f16) * B(32x16 f16) + C(16x16 f32)
  return __builtin_amdgcn_wmma_f32_16x16x32_f16(false, a, false, b, (short)0, c,
                                                false, false);
}

// Async 8-half (16B) global->LDS stage (ASYNCcnt); fallback: VGPR copy.
static __device__ __forceinline__ void stage_b128(_Float16* ldst,
                                                  const _Float16* gsrc) {
#if USE_ASYNC_LDS
  __builtin_amdgcn_global_load_async_to_lds_b128(
      (gv4i*)(unsigned long long)(size_t)gsrc,
      (lv4i*)(unsigned)(size_t)ldst, 0, 0);
#else
  *(v8h*)ldst = *(const v8h*)gsrc;
#endif
}

static __device__ __forceinline__ void wait_stage() {
#if USE_ASYNC_LDS
  __builtin_amdgcn_s_wait_asynccnt(0);
#endif
}

// Within a 32-wide K chunk, permute K index -> storage slot so that the ISA
// A-fragment (lanes 0-15: K0-7,K16-23 ; lanes 16-31: K8-15,K24-31) becomes a
// single contiguous 16-half read per lane:  [K0-7][K16-23][K8-15][K24-31]
static __device__ __forceinline__ int kperm(int t) {
  return (t & 7) | ((t & 8) << 1) | ((t & 16) >> 1);
}

// 16 contiguous halves -> v16h (compiles to 2x b128 / ds_load_b128)
static __device__ __forceinline__ v16h frag_row16(const _Float16* p) {
  v16h x;
#pragma unroll
  for (int h = 0; h < 16; ++h) x[h] = p[h];
  return x;
}

// Fragment from a 16-row x 32-half tile (row stride 32). Works for both the
// permuted A tile and the identically K-permuted B (W) tile.
static __device__ __forceinline__ v16h frag_tile16(const _Float16* base) {
  const int lane = threadIdx.x & 31;
  return frag_row16(base + (lane & 15) * 32 + (lane >> 4) * 16);
}

// ---------------------------------------------------------------------------
// f32 -> f16 convert (weights; resident in 192MB L2 afterwards)
// ---------------------------------------------------------------------------
__global__ void f32_to_f16_kernel(const float* __restrict__ in,
                                  _Float16* __restrict__ out, size_t n) {
  size_t i = (size_t)blockIdx.x * blockDim.x + threadIdx.x;
  const size_t stride = (size_t)gridDim.x * blockDim.x;
  for (; i < n; i += stride) out[i] = (_Float16)in[i];
}

// ---------------------------------------------------------------------------
// LayerNorm (f32 in) -> f16 out, one block per row
// ---------------------------------------------------------------------------
__global__ __launch_bounds__(256) void layernorm_to_f16_kernel(
    const float* __restrict__ x, const float* __restrict__ w,
    const float* __restrict__ b, _Float16* __restrict__ out) {
  __shared__ float red[256];
  const int row = blockIdx.x;
  const float* xr = x + (size_t)row * DIM;
  float s = 0.f;
  for (int i = threadIdx.x; i < DIM; i += 256) s += xr[i];
  red[threadIdx.x] = s; __syncthreads();
  for (int o = 128; o > 0; o >>= 1) {
    if (threadIdx.x < o) red[threadIdx.x] += red[threadIdx.x + o];
    __syncthreads();
  }
  const float mu = red[0] / DIM;
  __syncthreads();
  float vs = 0.f;
  for (int i = threadIdx.x; i < DIM; i += 256) { float d = xr[i] - mu; vs += d * d; }
  red[threadIdx.x] = vs; __syncthreads();
  for (int o = 128; o > 0; o >>= 1) {
    if (threadIdx.x < o) red[threadIdx.x] += red[threadIdx.x + o];
    __syncthreads();
  }
  const float rstd = rsqrtf(red[0] / DIM + EPS);
  for (int i = threadIdx.x; i < DIM; i += 256)
    out[(size_t)row * DIM + i] = (_Float16)((xr[i] - mu) * rstd * w[i] + b[i]);
}

// ---------------------------------------------------------------------------
// Tiled WMMA GEMM:  C[M,N] = A[M,K](f16) * W[N,K]^T(f16) + bias
//   BM=BN=128, BK=32; 256 threads = 8 waves (4 along M x 2 along N);
//   each wave owns a 32x64 sub-tile = 2x4 frags = 8 WMMA per K-step.
//   Double-buffered async global->LDS staging; A tile stored K-permuted so
//   every fragment is one contiguous 16-half read.
// ---------------------------------------------------------------------------
template <bool SILU, bool RES, bool OUT16>
__global__ __launch_bounds__(256) void gemm_wmma_kernel(
    const _Float16* __restrict__ A, const _Float16* __restrict__ W,
    const float* __restrict__ bias, const float* __restrict__ resid,
    float* __restrict__ Cf, _Float16* __restrict__ Ch, int M, int N, int K) {
  __shared__ __attribute__((aligned(16))) _Float16 sA[2][BM * BK];
  __shared__ __attribute__((aligned(16))) _Float16 sB[2][BN * BK];
  const int m0 = blockIdx.y * BM;
  const int n0 = blockIdx.x * BN;
  const int tid = threadIdx.x;
  const int wave = tid >> 5;
  const int wm = (wave & 3) * 32;   // wave row strip (2 x 16)
  const int wn = (wave >> 2) * 64;  // wave col strip (4 x 16)

  // Staging: 512 chunks of 8 halves per tile; each thread moves 2 chunks of
  // A and 2 of B. A chunks land in permuted slots ([K0-7][K16-23][K8-15][K24-31]).
  auto stage = [&](int buf, int kt) {
#pragma unroll
    for (int cc = 0; cc < 2; ++cc) {
      const int ch = tid + cc * 256;
      const int r = ch >> 2;          // 0..127
      const int sub = ch & 3;         // 8-half group within K row
      const int aslot = (sub == 0) ? 0 : (sub == 1) ? 16 : (sub == 2) ? 8 : 24;
      stage_b128(&sA[buf][r * 32 + aslot],
                 A + (size_t)(m0 + r) * K + kt * BK + sub * 8);
      stage_b128(&sB[buf][r * 32 + aslot],
                 W + (size_t)(n0 + r) * K + kt * BK + sub * 8);
    }
  };

  v8f acc[2][4];
  const v8f vzero = {};
#pragma unroll
  for (int i = 0; i < 2; ++i)
#pragma unroll
    for (int j = 0; j < 4; ++j) acc[i][j] = vzero;

  stage(0, 0);   // preload tile 0
  wait_stage();
  __syncthreads();

  const int nk = K / BK;
  for (int kt = 0; kt < nk; ++kt) {
    const int cur = kt & 1;
    if (kt + 1 < nk) stage(cur ^ 1, kt + 1);  // async prefetch next K-tile
    v16h a[2], b[4];
#pragma unroll
    for (int i = 0; i < 2; ++i) a[i] = frag_tile16(&sA[cur][(wm + i * 16) * 32]);
#pragma unroll
    for (int j = 0; j < 4; ++j) b[j] = frag_tile16(&sB[cur][(wn + j * 16) * 32]);
#pragma unroll
    for (int i = 0; i < 2; ++i)
#pragma unroll
      for (int j = 0; j < 4; ++j) acc[i][j] = wmma_f16(a[i], b[j], acc[i][j]);
    wait_stage();
    __syncthreads();
  }

  // Epilogue: C/D layout — lane (n) x VGPR (m), lanes 16..31 hold rows 8..15
  const int lane = tid & 31;
  const int cn = lane & 15;
  const int rbase = (lane < 16) ? 0 : 8;
#pragma unroll
  for (int i = 0; i < 2; ++i)
#pragma unroll
    for (int r = 0; r < 8; ++r) {
      const int gm = m0 + wm + i * 16 + rbase + r;
#pragma unroll
      for (int j = 0; j < 4; ++j) {
        const int gn = n0 + wn + j * 16 + cn;
        float v = acc[i][j][r] + bias[gn];
        if (SILU) v = v / (1.f + __expf(-v));
        if (RES) v += resid[(size_t)gm * N + gn];
        if (OUT16) Ch[(size_t)gm * N + gn] = (_Float16)v;
        else       Cf[(size_t)gm * N + gn] = v;
      }
    }
}

// ---------------------------------------------------------------------------
// RoPE + split qkv(f32,[S][3][H][80]):
//   qh: [H][S][96] f16, head-dim K-permuted per 32-chunk (A-frag friendly)
//   kh: [H][S][96] f16, head-dim K-permuted per 32-chunk (B-frag friendly), pad 0
//   vh: [H][96][S] f16, TRANSPOSED (keys contiguous for PV B-frags)
// ---------------------------------------------------------------------------
__global__ __launch_bounds__(256) void rope_split_kernel(
    const float* __restrict__ qkv, const float* __restrict__ rot,
    _Float16* __restrict__ qh, _Float16* __restrict__ kh,
    _Float16* __restrict__ vh) {
  const int s = blockIdx.x;
  const float* qr = qkv + (size_t)s * 3 * DIM;
  const float* fr = rot + (size_t)s * (HEAD_DIM / 2);
  for (int idx = threadIdx.x; idx < HEADS * HDP; idx += 256) {
    const int head = idx / HDP;
    const int d = idx % HDP;
    const int dperm = (d & ~31) | kperm(d & 31);
    const size_t oq = ((size_t)head * S_LEN + s) * HDP + dperm;
    const size_t ok = ((size_t)head * S_LEN + s) * HDP + dperm;
    if (d >= HEAD_DIM) {
      qh[oq] = (_Float16)0.f; kh[ok] = (_Float16)0.f;
      continue;
    }
    const float f = fr[d % (HEAD_DIM / 2)];
    const float c = __cosf(f), sn = __sinf(f);
    const float q = qr[head * HEAD_DIM + d];
    const float k = qr[DIM + head * HEAD_DIM + d];
    const float v = qr[2 * DIM + head * HEAD_DIM + d];
    const int d2 = (d < HEAD_DIM / 2) ? d + HEAD_DIM / 2 : d - HEAD_DIM / 2;
    const float sgn = (d < HEAD_DIM / 2) ? -1.f : 1.f;
    const float qrh = sgn * qr[head * HEAD_DIM + d2];
    const float krh = sgn * qr[DIM + head * HEAD_DIM + d2];
    qh[oq] = (_Float16)(q * c + qrh * sn);
    kh[ok] = (_Float16)(k * c + krh * sn);
    vh[((size_t)head * HDP + d) * S_LEN + s] = (_Float16)v;
  }
}

// ---------------------------------------------------------------------------
// Flash-style segmented attention. One wave per (16-query tile, head).
// 32 keys per iteration: QK^T = 6 wmma, PV = 5 full-K wmma.
// All fragments are contiguous 16-half reads (2x b128).
// ---------------------------------------------------------------------------
__global__ __launch_bounds__(32) void attn_wmma_kernel(
    const _Float16* __restrict__ qh, const _Float16* __restrict__ kh,
    const _Float16* __restrict__ vh, const int* __restrict__ cu, int nseg,
    _Float16* __restrict__ attn) {
  __shared__ __attribute__((aligned(16))) float    s_s[16 * 32];
  __shared__ __attribute__((aligned(16))) _Float16 s_p[16 * 32];
  __shared__ float s_scale[16];
  __shared__ float s_l[16];
  const int q0 = blockIdx.x * 16;
  const int head = blockIdx.y;
  int kbeg = 0, kend = S_LEN;
  for (int i = 0; i < nseg; ++i)
    if (q0 >= cu[i] && q0 < cu[i + 1]) { kbeg = cu[i]; kend = cu[i + 1]; }

  const int lane = threadIdx.x & 31;
  const int cn = lane & 15;
  const int half16 = (lane >> 4) * 16;
  const int rbase = (lane < 16) ? 0 : 8;
  const size_t hbase = (size_t)head * S_LEN * HDP;
  const float scale = 0.111803398875f;  // 1/sqrt(80)

  // Q fragments: 3 chunks of 16x32 (qh is K-permuted -> contiguous reads)
  v16h aq[3];
  {
    const _Float16* qrow = qh + hbase + (size_t)(q0 + cn) * HDP + half16;
#pragma unroll
    for (int c = 0; c < 3; ++c) aq[c] = frag_row16(qrow + c * 32);
  }

  v8f o[5];
  const v8f vzero = {};
#pragma unroll
  for (int f = 0; f < 5; ++f) o[f] = vzero;
  float m_i = -3.0e38f, l_i = 0.f;

  for (int k0 = kbeg; k0 < kend; k0 += 32) {
    // ---- scores(16x32) = Q * K^T ----
    v8f sc0 = vzero, sc1 = vzero;
#pragma unroll
    for (int c = 0; c < 3; ++c) {
      const _Float16* kr0 =
          kh + hbase + (size_t)(k0 + cn) * HDP + c * 32 + half16;
      sc0 = wmma_f16(aq[c], frag_row16(kr0), sc0);
      sc1 = wmma_f16(aq[c], frag_row16(kr0 + 16 * HDP), sc1);
    }
#pragma unroll
    for (int r = 0; r < 8; ++r) {
      s_s[(rbase + r) * 32 + cn]      = sc0[r];
      s_s[(rbase + r) * 32 + 16 + cn] = sc1[r];
    }
    asm volatile("s_wait_dscnt 0" ::: "memory");
    __builtin_amdgcn_wave_barrier();

    // ---- online softmax, one row per lane 0..15; write P natural order ----
    if (lane < 16) {
      float mx = -3.0e38f;
#pragma unroll
      for (int j = 0; j < 32; ++j) mx = fmaxf(mx, s_s[lane * 32 + j] * scale);
      const float nm = fmaxf(m_i, mx);
      const float corr = __expf(m_i - nm);
      float sum = 0.f;
#pragma unroll
      for (int j = 0; j < 32; ++j) {
        const float p = __expf(s_s[lane * 32 + j] * scale - nm);
        sum += p;
        s_p[lane * 32 + j] = (_Float16)p;
      }
      l_i = l_i * corr + sum;
      m_i = nm;
      s_scale[lane] = corr;
    }
    asm volatile("s_wait_dscnt 0" ::: "memory");
    __builtin_amdgcn_wave_barrier();

    // ---- rescale O, then O += P(16x32) @ V(32x80) ----
#pragma unroll
    for (int f = 0; f < 5; ++f)
#pragma unroll
      for (int r = 0; r < 8; ++r) o[f][r] *= s_scale[rbase + r];

    const v16h ap = frag_row16(s_p + cn * 32 + half16);  // P A-frag (natural)
#pragma unroll
    for (int f = 0; f < 5; ++f) {
      const _Float16* vcol =
          vh + ((size_t)head * HDP + f * 16 + cn) * S_LEN + k0 + half16;
      o[f] = wmma_f16(ap, frag_row16(vcol), o[f]);
    }
  }

  if (lane < 16) s_l[lane] = l_i;
  asm volatile("s_wait_dscnt 0" ::: "memory");
  __builtin_amdgcn_wave_barrier();
#pragma unroll
  for (int r = 0; r < 8; ++r) {
    const float inv = 1.f / s_l[rbase + r];
#pragma unroll
    for (int f = 0; f < 5; ++f)
      attn[(size_t)(q0 + rbase + r) * DIM + head * HEAD_DIM + f * 16 + cn] =
          (_Float16)(o[f][r] * inv);
  }
}

// ---------------------------------------------------------------------------
// Host-side launch sequence
// ---------------------------------------------------------------------------
extern "C" void kernel_launch(void* const* d_in, const int* in_sizes, int n_in,
                              void* d_out, int out_size, void* d_ws,
                              size_t ws_size, hipStream_t stream) {
  (void)n_in; (void)out_size; (void)ws_size;
  const float* hs     = (const float*)d_in[0];
  const float* rot    = (const float*)d_in[1];
  const float* ln1_w  = (const float*)d_in[2];
  const float* ln1_b  = (const float*)d_in[3];
  const float* ln2_w  = (const float*)d_in[4];
  const float* ln2_b  = (const float*)d_in[5];
  const float* qkv_w  = (const float*)d_in[6];
  const float* qkv_b  = (const float*)d_in[7];
  const float* proj_w = (const float*)d_in[8];
  const float* proj_b = (const float*)d_in[9];
  const float* fc1_w  = (const float*)d_in[10];
  const float* fc1_b  = (const float*)d_in[11];
  const float* fc2_w  = (const float*)d_in[12];
  const float* fc2_b  = (const float*)d_in[13];
  const int*   cu     = (const int*)d_in[14];
  const int nseg = in_sizes[14] - 1;
  float* out = (float*)d_out;

  char* ws = (char*)d_ws;
  size_t off = 0;
  auto alloc = [&](size_t bytes) -> void* {
    void* p = ws + off;
    off = (off + bytes + 255) & ~(size_t)255;
    return p;
  };
  _Float16* w_qkv16  = (_Float16*)alloc((size_t)3 * DIM * DIM * 2);
  _Float16* w_proj16 = (_Float16*)alloc((size_t)DIM * DIM * 2);
  _Float16* w_fc116  = (_Float16*)alloc((size_t)MLP_HIDDEN * DIM * 2);
  _Float16* w_fc216  = (_Float16*)alloc((size_t)DIM * MLP_HIDDEN * 2);
  _Float16* xh       = (_Float16*)alloc((size_t)S_LEN * DIM * 2);
  float*    qkv_f    = (float*)   alloc((size_t)S_LEN * 3 * DIM * 4);
  _Float16* qhp      = (_Float16*)alloc((size_t)HEADS * S_LEN * HDP * 2);
  _Float16* khp      = (_Float16*)alloc((size_t)HEADS * S_LEN * HDP * 2);
  _Float16* vhp      = (_Float16*)alloc((size_t)HEADS * S_LEN * HDP * 2);
  _Float16* attn16   = (_Float16*)alloc((size_t)S_LEN * DIM * 2);
  float*    h_f      = (float*)   alloc((size_t)S_LEN * DIM * 4);
  _Float16* yh       = (_Float16*)alloc((size_t)S_LEN * DIM * 2);
  _Float16* a1h      = (_Float16*)alloc((size_t)S_LEN * MLP_HIDDEN * 2);

  // 1) weights -> f16
  f32_to_f16_kernel<<<2048, 256, 0, stream>>>(qkv_w,  w_qkv16,  (size_t)3 * DIM * DIM);
  f32_to_f16_kernel<<<2048, 256, 0, stream>>>(proj_w, w_proj16, (size_t)DIM * DIM);
  f32_to_f16_kernel<<<2048, 256, 0, stream>>>(fc1_w,  w_fc116,  (size_t)MLP_HIDDEN * DIM);
  f32_to_f16_kernel<<<2048, 256, 0, stream>>>(fc2_w,  w_fc216,  (size_t)DIM * MLP_HIDDEN);

  // 2) LN1 -> f16
  layernorm_to_f16_kernel<<<S_LEN, 256, 0, stream>>>(hs, ln1_w, ln1_b, xh);

  // 3) QKV GEMM (f32 out for RoPE)
  gemm_wmma_kernel<false, false, false>
      <<<dim3(3 * DIM / BN, S_LEN / BM), 256, 0, stream>>>(
          xh, w_qkv16, qkv_b, nullptr, qkv_f, nullptr, S_LEN, 3 * DIM, DIM);

  // 4) RoPE + head-major split (q permuted, k permuted, v transposed)
  rope_split_kernel<<<S_LEN, 256, 0, stream>>>(qkv_f, rot, qhp, khp, vhp);

  // 5) segmented flash attention
  attn_wmma_kernel<<<dim3(S_LEN / 16, HEADS), 32, 0, stream>>>(
      qhp, khp, vhp, cu, nseg, attn16);

  // 6) proj GEMM + residual -> h (f32)
  gemm_wmma_kernel<false, true, false>
      <<<dim3(DIM / BN, S_LEN / BM), 256, 0, stream>>>(
          attn16, w_proj16, proj_b, hs, h_f, nullptr, S_LEN, DIM, DIM);

  // 7) LN2 -> f16
  layernorm_to_f16_kernel<<<S_LEN, 256, 0, stream>>>(h_f, ln2_w, ln2_b, yh);

  // 8) FC1 + SiLU -> f16
  gemm_wmma_kernel<true, false, true>
      <<<dim3(MLP_HIDDEN / BN, S_LEN / BM), 256, 0, stream>>>(
          yh, w_fc116, fc1_b, nullptr, nullptr, a1h, S_LEN, MLP_HIDDEN, DIM);

  // 9) FC2 + residual -> out (f32)
  gemm_wmma_kernel<false, true, false>
      <<<dim3(DIM / BN, S_LEN / BM), 256, 0, stream>>>(
          a1h, w_fc216, fc2_b, h_f, out, nullptr, S_LEN, DIM, MLP_HIDDEN);
}
